// AFNO_80736795230698
// MI455X (gfx1250) — matrix operand order repaired and checked by reading, not compiled, not run on hardware
//
#include <hip/hip_runtime.h>

// ---------------- CDNA5 WMMA types ----------------
typedef __attribute__((ext_vector_type(16))) __bf16       v16bf;
typedef __attribute__((ext_vector_type(8)))  __bf16       v8bf;
typedef __attribute__((ext_vector_type(8)))  float        v8f;
typedef __attribute__((ext_vector_type(8)))  unsigned int v8ui;

#define PI_F 3.14159265358979323846f

// AFNO2D problem constants
#define BATCH 16
#define HDIM  64
#define WDIM  64
#define CDIM  448
#define WFREQ 33          // 64/2 + 1
#define NBLK  8
#define BSZ   56
#define LAMBDA_SS 0.01f

// LDS row stride (bf16 elements): rows stay 16B-aligned, 16 contiguous K per frag
#define LSTR 72

static __device__ __forceinline__ v8f wmma_bf16(v16bf a, v16bf b, v8f c) {
  return __builtin_amdgcn_wmma_f32_16x16x32_bf16(false, a, false, b, (short)0, c,
                                                 false, false);
}
static __device__ __forceinline__ v8f wmma2(v16bf ah, v16bf al, v16bf b, v8f c) {
  c = wmma_bf16(ah, b, c);
  c = wmma_bf16(al, b, c);
  return c;
}
static __device__ __forceinline__ v8f wmma3(v16bf ah, v16bf al, v16bf bh, v16bf bl, v8f c) {
  c = wmma_bf16(ah, bh, c);
  c = wmma_bf16(ah, bl, c);
  c = wmma_bf16(al, bh, c);
  return c;
}

static __device__ __forceinline__ void splitf(float v, __bf16 &h, __bf16 &l) {
  __bf16 hh = (__bf16)v;
  h = hh;
  l = (__bf16)(v - (float)hh);
}

// 16-bit A-matrix (16x32) element->K mapping (contiguous: k = 16*laneHi + e)
static __device__ __forceinline__ int a_kidx(int e, int laneHi) {
  return (laneHi << 4) + e;
}

// concat two 16B LDS vectors into one fragment
static __device__ __forceinline__ v16bf cat2(v8bf a, v8bf b) {
  return __builtin_shufflevector(a, b, 0,1,2,3,4,5,6,7,8,9,10,11,12,13,14,15);
}
// B fragment: 16 contiguous bf16 at row-major [col][k0..k0+15]
static __device__ __forceinline__ v16bf ldB(const __bf16* p) {
  return cat2(*(const v8bf*)p, *(const v8bf*)(p + 8));
}
// A fragment from LDS [row][k]: 16 contiguous k0..k0+15
static __device__ __forceinline__ v16bf ldA(const __bf16* p) {
  return cat2(*(const v8bf*)p, *(const v8bf*)(p + 8));
}
// packed bf16 sign flip (negate a whole fragment, 8 VALU xors)
static __device__ __forceinline__ v16bf negbf(v16bf x) {
  v8ui u = __builtin_bit_cast(v8ui, x);
  u ^= 0x80008000u;
  return __builtin_bit_cast(v16bf, u);
}

// ============================================================================
// Table builder: DFT twiddles pre-baked in WMMA A-fragment layout (v16bf/lane).
// Angles are exact multiples of 2*pi/64 and results are split to bf16 hi/lo
// (~2^-17), so hardware-native v_sin/v_cos (__sinf/__cosf) is precision-
// adequate and keeps this kernel a few dozen VALU ops instead of inlined libm.
// Layout (v16bf units):
//   stage1 (row rfft, 3 mt x 2 ks):  [0]    cH, [192] cL, [384] sH, [576] sL
//   stage2 (col fft, 4 mt x 2 ks):   [768]  cH,cL,sinP H/L, sinN H/L (x256 each)
//   stage5 (row irfft, 4 mt x 2 ks): [2304] crH,crL,ciH,ciL (x256 each)
// ============================================================================
__global__ __launch_bounds__(32) void afno_build_tables(v16bf* __restrict__ tab) {
  const int blk = blockIdx.x;
  const int lane = threadIdx.x;
  const int laneHi = lane >> 4, mlane = lane & 15;
  const float w0 = 2.0f * PI_F / 64.0f;

  if (blk < 6) {                       // stage 1: rfft rows, ortho 1/64 folded
    const int mt = blk >> 1, ks = blk & 1;
    const int m = mt * 16 + mlane;
    v16bf cH, cL, sH, sL;
    #pragma unroll
    for (int e = 0; e < 16; ++e) {
      int k = ks * 32 + a_kidx(e, laneHi);
      float ang = w0 * (float)((m * k) & 63);
      float cv =  __cosf(ang) * (1.0f / 64.0f);
      float sv = -__sinf(ang) * (1.0f / 64.0f);
      __bf16 h, l;
      splitf(cv, h, l); cH[e] = h; cL[e] = l;
      splitf(sv, h, l); sH[e] = h; sL[e] = l;
    }
    int idx = (mt * 2 + ks) * 32 + lane;
    tab[idx] = cH; tab[192 + idx] = cL; tab[384 + idx] = sH; tab[576 + idx] = sL;
  } else if (blk < 14) {               // stage 2/4: 64-pt complex FFT columns
    const int t = blk - 6;
    const int mt = t >> 1, ks = t & 1;
    const int m = mt * 16 + mlane;
    v16bf cH, cL, pH, pL, nH, nL;
    #pragma unroll
    for (int e = 0; e < 16; ++e) {
      int k = ks * 32 + a_kidx(e, laneHi);
      float ang = w0 * (float)((m * k) & 63);
      float cv = __cosf(ang);
      float sv = __sinf(ang);
      __bf16 h, l;
      splitf(cv, h, l);  cH[e] = h; cL[e] = l;
      splitf(sv, h, l);  pH[e] = h; pL[e] = l;
      splitf(-sv, h, l); nH[e] = h; nL[e] = l;
    }
    int idx = 768 + (mt * 2 + ks) * 32 + lane;
    tab[idx] = cH;        tab[idx + 256] = cL;
    tab[idx + 512] = pH;  tab[idx + 768] = pL;
    tab[idx + 1024] = nH; tab[idx + 1280] = nL;
  } else {                             // stage 5: irfft (Hermitian fold + 1/64)
    const int t = blk - 14;
    const int mt = t >> 1, ks = t & 1;
    const int m = mt * 16 + mlane;
    v16bf rH, rL, iH, iL;
    #pragma unroll
    for (int e = 0; e < 16; ++e) {
      int k = ks * 32 + a_kidx(e, laneHi);
      float cv = 0.f, sv = 0.f;
      if (k < WFREQ) {
        float alpha = (k == 0 || k == 32) ? 1.0f : 2.0f;
        float ang = w0 * (float)((m * k) & 63);
        cv =  alpha * __cosf(ang) * (1.0f / 64.0f);
        sv = -alpha * __sinf(ang) * (1.0f / 64.0f);
      }
      __bf16 h, l;
      splitf(cv, h, l); rH[e] = h; rL[e] = l;
      splitf(sv, h, l); iH[e] = h; iL[e] = l;
    }
    int idx = 2304 + (mt * 2 + ks) * 32 + lane;
    tab[idx] = rH;       tab[idx + 256] = rL;
    tab[idx + 512] = iH; tab[idx + 768] = iL;
  }
}

// ============================================================================
// Stage 1: rfft along W as GEMM  Xf[33 x 448] = DFT[33 x 64] * x[64 x 448]
// ============================================================================
__global__ __launch_bounds__(256) void afno_row_rfft(
    const float* __restrict__ x, float* __restrict__ Xr, float* __restrict__ Xi,
    const v16bf* __restrict__ tcH, const v16bf* __restrict__ tcL,
    const v16bf* __restrict__ tsH, const v16bf* __restrict__ tsL) {
  __shared__ __align__(16) __bf16 Bh[64][LSTR];
  __shared__ __align__(16) __bf16 Bl[64][LSTR];
  const int b  = blockIdx.x >> 6;
  const int h  = blockIdx.x & 63;
  const int c0 = blockIdx.y * 64;
  // warm the shared twiddle tables into GL2 (global_prefetch_b8)
  __builtin_prefetch(tcH + (threadIdx.x & 31), 0, 3);
  __builtin_prefetch(tsH + (threadIdx.x & 31), 0, 3);
  const float* src = x + ((size_t)b * (HDIM * WDIM) + (size_t)h * WDIM) * CDIM + c0;
  for (int i = threadIdx.x; i < 64 * 64; i += 256) {
    int w = i >> 6, cc = i & 63;
    __bf16 hh, ll;
    splitf(src[(size_t)w * CDIM + cc], hh, ll);
    Bh[cc][w] = hh;   // transposed: [col][k]
    Bl[cc][w] = ll;
  }
  __syncthreads();
  const int wave = threadIdx.x >> 5;
  const int lane = threadIdx.x & 31;
  const int laneHi = lane >> 4, mlane = lane & 15;
  for (int tile = wave; tile < 12; tile += 8) {    // 3 M-tiles x 4 N-tiles
    const int mt = tile >> 2, nt = tile & 3;
    const int col = nt * 16 + mlane;
    v8f accR = {}, accI = {};
    #pragma unroll
    for (int ks = 0; ks < 2; ++ks) {
      const int aidx = (mt * 2 + ks) * 32 + lane;
      const int k0 = ks * 32 + laneHi * 16;
      v16bf bh = ldB(&Bh[col][k0]);
      v16bf bl = ldB(&Bl[col][k0]);
      accR = wmma3(tcH[aidx], tcL[aidx], bh, bl, accR);
      accI = wmma3(tsH[aidx], tsL[aidx], bh, bl, accI);
    }
    #pragma unroll
    for (int r = 0; r < 8; ++r) {
      int mm = mt * 16 + (laneHi << 3) + r;
      if (mm < WFREQ) {
        size_t off = (((size_t)b * HDIM + h) * WFREQ + mm) * CDIM + c0 + col;
        Xr[off] = accR[r];
        Xi[off] = accI[r];
      }
    }
  }
}

// ============================================================================
// Stages 2 & 4: complex 64-pt FFT along H.  Sin tables are swapped by the host
// for the inverse transform (sinP <-> sinN), so one kernel serves both.
//   Re = C*Br + S*Bi ; Im = C*Bi + (-S)*Br
// ============================================================================
__global__ __launch_bounds__(256) void afno_col_fft(
    const float* __restrict__ Sr, const float* __restrict__ Si,
    float* __restrict__ Dr, float* __restrict__ Di,
    const v16bf* __restrict__ tcH, const v16bf* __restrict__ tcL,
    const v16bf* __restrict__ tsH, const v16bf* __restrict__ tsL,
    const v16bf* __restrict__ tnH, const v16bf* __restrict__ tnL) {
  __shared__ __align__(16) __bf16 Rh[64][LSTR];
  __shared__ __align__(16) __bf16 Rl[64][LSTR];
  __shared__ __align__(16) __bf16 Ih[64][LSTR];
  __shared__ __align__(16) __bf16 Il[64][LSTR];
  const int b  = blockIdx.x / WFREQ;
  const int wf = blockIdx.x % WFREQ;
  const int c0 = blockIdx.y * 64;
  __builtin_prefetch(tcH + (threadIdx.x & 31), 0, 3);
  __builtin_prefetch(tsH + (threadIdx.x & 31), 0, 3);
  const size_t hstride = (size_t)WFREQ * CDIM;
  const size_t base = ((size_t)b * HDIM * WFREQ + wf) * CDIM + c0;
  for (int i = threadIdx.x; i < 64 * 64; i += 256) {
    int hh = i >> 6, cc = i & 63;
    __bf16 h, l;
    splitf(Sr[base + (size_t)hh * hstride + cc], h, l);
    Rh[cc][hh] = h; Rl[cc][hh] = l;
    splitf(Si[base + (size_t)hh * hstride + cc], h, l);
    Ih[cc][hh] = h; Il[cc][hh] = l;
  }
  __syncthreads();
  const int wave = threadIdx.x >> 5;
  const int lane = threadIdx.x & 31;
  const int laneHi = lane >> 4, mlane = lane & 15;
  for (int tile = wave; tile < 16; tile += 8) {    // 4 M-tiles x 4 N-tiles
    const int mt = tile >> 2, nt = tile & 3;
    const int col = nt * 16 + mlane;
    v8f accR = {}, accI = {};
    #pragma unroll
    for (int ks = 0; ks < 2; ++ks) {
      const int aidx = (mt * 2 + ks) * 32 + lane;
      const int k0 = ks * 32 + laneHi * 16;
      v16bf brh = ldB(&Rh[col][k0]);
      v16bf brl = ldB(&Rl[col][k0]);
      v16bf bih = ldB(&Ih[col][k0]);
      v16bf bil = ldB(&Il[col][k0]);
      accR = wmma3(tcH[aidx], tcL[aidx], brh, brl, accR);
      accR = wmma3(tsH[aidx], tsL[aidx], bih, bil, accR);
      accI = wmma3(tcH[aidx], tcL[aidx], bih, bil, accI);
      accI = wmma3(tnH[aidx], tnL[aidx], brh, brl, accI);
    }
    #pragma unroll
    for (int r = 0; r < 8; ++r) {
      int mm = mt * 16 + (laneHi << 3) + r;
      size_t off = base + (size_t)mm * hstride + col;
      Dr[off] = accR[r];
      Di[off] = accI[r];
    }
  }
}

// ============================================================================
// Stage 3: block-diagonal complex MLP (relu, softshrink); one wave32 / block.
// Weights transposed in LDS (fragment-contiguous bf16); activations pre-split.
// ============================================================================
__global__ __launch_bounds__(32) void afno_block_mlp(
    const float* __restrict__ Xr, const float* __restrict__ Xi,
    const float* __restrict__ w1, const float* __restrict__ b1,
    const float* __restrict__ w2, const float* __restrict__ b2,
    float* __restrict__ Yr, float* __restrict__ Yi) {
  __shared__ __align__(16) __bf16 W1r[64][LSTR], W1i[64][LSTR];
  __shared__ __align__(16) __bf16 W2r[64][LSTR], W2i[64][LSTR];
  __shared__ __align__(16) __bf16 XrH[16][LSTR], XrL[16][LSTR];
  __shared__ __align__(16) __bf16 XiH[16][LSTR], XiL[16][LSTR];
  __shared__ __align__(16) __bf16 O1rH[16][LSTR], O1rL[16][LSTR];
  __shared__ __align__(16) __bf16 O1iH[16][LSTR], O1iL[16][LSTR];
  const int g = blockIdx.x;   // group of 16 frequency points
  const int n = blockIdx.y;   // feature block
  const int lane = threadIdx.x;
  const int laneHi = lane >> 4, mlane = lane & 15;

  // weights -> LDS, transposed [out col][in k], zero padded 56->64
  #pragma unroll 4
  for (int i = lane; i < 64 * 64; i += 32) {
    int kk = i >> 6, oo = i & 63;
    float v1r = 0.f, v1i = 0.f, v2r = 0.f, v2i = 0.f;
    if (kk < BSZ && oo < BSZ) {
      size_t o = ((size_t)n * BSZ + kk) * BSZ + oo;
      v1r = w1[o];  v1i = w1[(size_t)NBLK * BSZ * BSZ + o];
      v2r = w2[o];  v2i = w2[(size_t)NBLK * BSZ * BSZ + o];
    }
    W1r[oo][kk] = (__bf16)v1r; W1i[oo][kk] = (__bf16)v1i;
    W2r[oo][kk] = (__bf16)v2r; W2i[oo][kk] = (__bf16)v2i;
  }
  // activations -> LDS, pre-split hi/lo, zero padded
  #pragma unroll 4
  for (int i = lane; i < 16 * 64; i += 32) {
    int row = i >> 6, k = i & 63;
    size_t prow = ((size_t)g * 16 + row) * CDIM + (size_t)n * BSZ;
    float vr = (k < BSZ) ? Xr[prow + k] : 0.0f;
    float vi = (k < BSZ) ? Xi[prow + k] : 0.0f;
    __bf16 h, l;
    splitf(vr, h, l); XrH[row][k] = h; XrL[row][k] = l;
    splitf(vi, h, l); XiH[row][k] = h; XiL[row][k] = l;
  }
  __syncthreads();

  // layer-1 A fragments
  v16bf xrh[2], xrl[2], xih[2], xil[2], nih[2], nil_[2];
  #pragma unroll
  for (int ks = 0; ks < 2; ++ks) {
    const int k0 = ks * 32 + laneHi * 16;
    xrh[ks] = ldA(&XrH[mlane][k0]);
    xrl[ks] = ldA(&XrL[mlane][k0]);
    xih[ks] = ldA(&XiH[mlane][k0]);
    xil[ks] = ldA(&XiL[mlane][k0]);
    nih[ks] = negbf(xih[ks]);
    nil_[ks] = negbf(xil[ks]);
  }

  // layer 1: o1 = relu(x (*) W1 + b1)
  #pragma unroll
  for (int nt = 0; nt < 4; ++nt) {
    const int col = nt * 16 + mlane;
    float biasR = (col < BSZ) ? b1[(size_t)n * BSZ + col] : 0.0f;
    float biasI = (col < BSZ) ? b1[(size_t)NBLK * BSZ + (size_t)n * BSZ + col] : 0.0f;
    v8f aR, aI;
    #pragma unroll
    for (int r = 0; r < 8; ++r) { aR[r] = biasR; aI[r] = biasI; }
    #pragma unroll
    for (int ks = 0; ks < 2; ++ks) {
      const int k0 = ks * 32 + laneHi * 16;
      v16bf w1rF = ldB(&W1r[col][k0]);
      v16bf w1iF = ldB(&W1i[col][k0]);
      aR = wmma2(xrh[ks], xrl[ks], w1rF, aR);    // + xr*W1r
      aR = wmma2(nih[ks], nil_[ks], w1iF, aR);   // - xi*W1i
      aI = wmma2(xih[ks], xil[ks], w1rF, aI);    // + xi*W1r
      aI = wmma2(xrh[ks], xrl[ks], w1iF, aI);    // + xr*W1i
    }
    #pragma unroll
    for (int r = 0; r < 8; ++r) {
      int row = (laneHi << 3) + r;
      __bf16 h, l;
      splitf(fmaxf(aR[r], 0.0f), h, l); O1rH[row][col] = h; O1rL[row][col] = l;
      splitf(fmaxf(aI[r], 0.0f), h, l); O1iH[row][col] = h; O1iL[row][col] = l;
    }
  }
  __syncthreads();

  // layer-2 A fragments
  v16bf orh[2], orl[2], oih[2], oil[2], noh[2], nol[2];
  #pragma unroll
  for (int ks = 0; ks < 2; ++ks) {
    const int k0 = ks * 32 + laneHi * 16;
    orh[ks] = ldA(&O1rH[mlane][k0]);
    orl[ks] = ldA(&O1rL[mlane][k0]);
    oih[ks] = ldA(&O1iH[mlane][k0]);
    oil[ks] = ldA(&O1iL[mlane][k0]);
    noh[ks] = negbf(oih[ks]);
    nol[ks] = negbf(oil[ks]);
  }

  // layer 2 + softshrink
  #pragma unroll
  for (int nt = 0; nt < 4; ++nt) {
    const int col = nt * 16 + mlane;
    float biasR = (col < BSZ) ? b2[(size_t)n * BSZ + col] : 0.0f;
    float biasI = (col < BSZ) ? b2[(size_t)NBLK * BSZ + (size_t)n * BSZ + col] : 0.0f;
    v8f aR, aI;
    #pragma unroll
    for (int r = 0; r < 8; ++r) { aR[r] = biasR; aI[r] = biasI; }
    #pragma unroll
    for (int ks = 0; ks < 2; ++ks) {
      const int k0 = ks * 32 + laneHi * 16;
      v16bf w2rF = ldB(&W2r[col][k0]);
      v16bf w2iF = ldB(&W2i[col][k0]);
      aR = wmma2(orh[ks], orl[ks], w2rF, aR);
      aR = wmma2(noh[ks], nol[ks], w2iF, aR);
      aI = wmma2(oih[ks], oil[ks], w2rF, aI);
      aI = wmma2(orh[ks], orl[ks], w2iF, aI);
    }
    if (col < BSZ) {
      #pragma unroll
      for (int r = 0; r < 8; ++r) {
        int row = (laneHi << 3) + r;
        size_t off = ((size_t)g * 16 + row) * CDIM + (size_t)n * BSZ + col;
        float vr = aR[r], vi = aI[r];
        float mr = fabsf(vr) - LAMBDA_SS;
        float mi = fabsf(vi) - LAMBDA_SS;
        Yr[off] = (mr > 0.0f) ? copysignf(mr, vr) : 0.0f;
        Yi[off] = (mi > 0.0f) ? copysignf(mi, vi) : 0.0f;
      }
    }
  }
}

// ============================================================================
// Stage 5: irfft along W (Hermitian fold baked into tables) + residual add.
// ============================================================================
__global__ __launch_bounds__(256) void afno_row_irfft(
    const float* __restrict__ Yr, const float* __restrict__ Yi,
    const float* __restrict__ x, float* __restrict__ out,
    const v16bf* __restrict__ trH, const v16bf* __restrict__ trL,
    const v16bf* __restrict__ tiH, const v16bf* __restrict__ tiL) {
  __shared__ __align__(16) __bf16 Rh[64][LSTR];
  __shared__ __align__(16) __bf16 Rl[64][LSTR];
  __shared__ __align__(16) __bf16 Ih[64][LSTR];
  __shared__ __align__(16) __bf16 Il[64][LSTR];
  const int b  = blockIdx.x >> 6;
  const int h  = blockIdx.x & 63;
  const int c0 = blockIdx.y * 64;
  __builtin_prefetch(trH + (threadIdx.x & 31), 0, 3);
  __builtin_prefetch(tiH + (threadIdx.x & 31), 0, 3);
  const size_t base = (((size_t)b * HDIM + h) * WFREQ) * CDIM + c0;
  for (int i = threadIdx.x; i < 64 * 64; i += 256) {
    int k = i >> 6, cc = i & 63;
    float vr = 0.f, vi = 0.f;
    if (k < WFREQ) {
      vr = Yr[base + (size_t)k * CDIM + cc];
      vi = Yi[base + (size_t)k * CDIM + cc];
    }
    __bf16 hh, ll;
    splitf(vr, hh, ll); Rh[cc][k] = hh; Rl[cc][k] = ll;
    splitf(vi, hh, ll); Ih[cc][k] = hh; Il[cc][k] = ll;
  }
  __syncthreads();
  const int wave = threadIdx.x >> 5;
  const int lane = threadIdx.x & 31;
  const int laneHi = lane >> 4, mlane = lane & 15;
  for (int tile = wave; tile < 16; tile += 8) {
    const int mt = tile >> 2, nt = tile & 3;
    const int col = nt * 16 + mlane;
    v8f acc = {};
    #pragma unroll
    for (int ks = 0; ks < 2; ++ks) {
      const int aidx = (mt * 2 + ks) * 32 + lane;
      const int k0 = ks * 32 + laneHi * 16;
      v16bf brh = ldB(&Rh[col][k0]);
      v16bf brl = ldB(&Rl[col][k0]);
      v16bf bih = ldB(&Ih[col][k0]);
      v16bf bil = ldB(&Il[col][k0]);
      acc = wmma3(trH[aidx], trL[aidx], brh, brl, acc);
      acc = wmma3(tiH[aidx], tiL[aidx], bih, bil, acc);
    }
    #pragma unroll
    for (int r = 0; r < 8; ++r) {
      int w = mt * 16 + (laneHi << 3) + r;
      size_t off = ((size_t)b * (HDIM * WDIM) + (size_t)h * WDIM + w) * CDIM + c0 + col;
      out[off] = acc[r] + x[off];
    }
  }
}

// ============================================================================
// Host-side launch.
// Workspace: 4 x (B*H*WFREQ*C) f32 spectral arrays (231 MiB) + 104 KB tables.
// ============================================================================
extern "C" void kernel_launch(void* const* d_in, const int* in_sizes, int n_in,
                              void* d_out, int out_size, void* d_ws, size_t ws_size,
                              hipStream_t stream) {
  (void)in_sizes; (void)n_in; (void)out_size; (void)ws_size;
  const float* x  = (const float*)d_in[0];
  const float* w1 = (const float*)d_in[1];
  const float* b1 = (const float*)d_in[2];
  const float* w2 = (const float*)d_in[3];
  const float* b2 = (const float*)d_in[4];
  float* out = (float*)d_out;

  const size_t SZ = (size_t)BATCH * HDIM * WFREQ * CDIM;  // 15,138,816 floats
  float* Xr = (float*)d_ws;
  float* Xi = Xr + SZ;
  float* Yr = Xi + SZ;
  float* Yi = Yr + SZ;
  v16bf* tab = (v16bf*)(Yi + SZ);

  const v16bf* t1 = tab;          // stage 1: 4 comps x 192
  const v16bf* t2 = tab + 768;    // stage 2: 6 comps x 256
  const v16bf* t5 = tab + 2304;   // stage 5: 4 comps x 256

  const int nchunks = CDIM / 64;  // 7

  // 0) twiddle fragment tables (22 waves of trig, L2 resident afterwards)
  afno_build_tables<<<22, 32, 0, stream>>>(tab);
  // 1) rfft along W
  afno_row_rfft<<<dim3(BATCH * HDIM, nchunks), 256, 0, stream>>>(
      x, Xr, Xi, t1, t1 + 192, t1 + 384, t1 + 576);
  // 2) forward complex FFT along H (in place): S = +sin, N = -sin
  afno_col_fft<<<dim3(BATCH * WFREQ, nchunks), 256, 0, stream>>>(
      Xr, Xi, Xr, Xi,
      t2, t2 + 256, t2 + 512, t2 + 768, t2 + 1024, t2 + 1280);
  // 3) block-diagonal complex MLP + relu + softshrink
  afno_block_mlp<<<dim3((BATCH * HDIM * WFREQ) / 16, NBLK), 32, 0, stream>>>(
      Xr, Xi, w1, b1, w2, b2, Yr, Yi);
  // 4) inverse complex FFT along H (in place): sin tables swapped
  afno_col_fft<<<dim3(BATCH * WFREQ, nchunks), 256, 0, stream>>>(
      Yr, Yi, Yr, Yi,
      t2, t2 + 256, t2 + 1024, t2 + 1280, t2 + 512, t2 + 768);
  // 5) irfft along W + residual
  afno_row_irfft<<<dim3(BATCH * HDIM, nchunks), 256, 0, stream>>>(
      Yr, Yi, x, out, t5, t5 + 256, t5 + 512, t5 + 768);
}
